// ASPPConv_37572373905799
// MI455X (gfx1250) — matrix operand order repaired and checked by reading, not compiled, not run on hardware
//
#include <hip/hip_runtime.h>
#include <hip/hip_bf16.h>

// ---------------- problem constants (from reference) ----------------
#define BATCH   8
#define CIN     256
#define COUT    256
#define LLEN    16384
#define KT      9
#define DIL     6
#define PADL    24          // DIL * (KT/2)
#define KDIM    (CIN*KT)    // 2304
#define LPB     64          // l-tile per workgroup
#define HALO    (LPB + 2*PADL)      // 112 staged positions
#define NTOT    (BATCH*COUT*LLEN)   // 33554432 output elements
#define INV2S2  (1.0f/72.0f)        // 1/(2*sigma^2), sigma = 6
#define XSTR    (CIN + 8)           // LDS row stride in shorts (528 B, bank-skewed)

typedef __attribute__((ext_vector_type(16))) __bf16 v16bf;
typedef __attribute__((ext_vector_type(8)))  float  v8f;

union FragA { v16bf v; uint4 q[2]; };
union FragB { v16bf v; uint4 q[2]; unsigned u[8]; };

__device__ __forceinline__ unsigned short f2bf(float f) {
    unsigned u = __builtin_bit_cast(unsigned, f);
    unsigned r = u + 0x7fffu + ((u >> 16) & 1u);   // round-to-nearest-even
    return (unsigned short)(r >> 16);
}

// ---------------- 1) weight repack: W[o][c][k] f32 -> Wbf[o][k*256+c] bf16 ----
__global__ __launch_bounds__(256) void prep_w(const float* __restrict__ w,
                                              unsigned short* __restrict__ wbf) {
    int idx = blockIdx.x * 256 + threadIdx.x;      // 0 .. 589823
    int o   = idx / KDIM;
    int rem = idx - o * KDIM;
    int k   = rem >> 8;
    int c   = rem & 255;
    wbf[idx] = f2bf(w[o * KDIM + c * KT + k]);
}

// ---------------- 2) main WMMA conv kernel ---------------------------------
__global__ __launch_bounds__(256) void conv_wmma(const float* __restrict__ x,
                                                 const float* __restrict__ coords,
                                                 const unsigned short* __restrict__ wbf,
                                                 float* __restrict__ y) {
    __shared__ unsigned short xs[HALO][XSTR];      // bf16 x halo tile (59136 B)
    __shared__ unsigned wkb[KT][LPB];              // Gaussian tap weights, bf16 pair-packed

    const int wg   = blockIdx.x;
    const int b    = wg >> 8;                      // 256 tiles per batch
    const int l0   = (wg & 255) * LPB;
    const int t    = threadIdx.x;
    const int lane = t & 31;
    const int wave = t >> 5;
    const int half = lane >> 4;                    // 0: lanes 0-15, 1: lanes 16-31
    const int nIn  = lane & 15;

    // ---- Gaussian tap weights for this tile (threads 0..63 -> one column) ----
    if (t < LPB) {
        int l = l0 + t;
        float cx = coords[(b * 3 + 0) * LLEN + l];
        float cy = coords[(b * 3 + 1) * LLEN + l];
        float cz = coords[(b * 3 + 2) * LLEN + l];
        #pragma unroll
        for (int k = 0; k < KT; ++k) {
            int lk = l + k * DIL - PADL;
            float ax = 0.f, ay = 0.f, az = 0.f;
            if (lk >= 0 && lk < LLEN) {
                ax = coords[(b * 3 + 0) * LLEN + lk];
                ay = coords[(b * 3 + 1) * LLEN + lk];
                az = coords[(b * 3 + 2) * LLEN + lk];
            }
            float dx = ax - cx, dy = ay - cy, dz = az - cz;
            float d2 = dx * dx + dy * dy + dz * dz;
            unsigned short wb = f2bf(__expf(-d2 * INV2S2));
            wkb[k][t] = ((unsigned)wb << 16) | wb;        // splat pair for pk-mul
        }
    }

    // ---- stage x halo tile once: xs[pos][c] = bf16(x[b][c][l0-24+pos]) ----
    {
        const int lsub = t & 31;
        const int cgrp = t >> 5;
        #pragma unroll 4
        for (int i = 0; i < 32; ++i) {
            int c = cgrp + 8 * i;
            const float* xr = x + (b * CIN + c) * LLEN + (l0 - PADL);
            #pragma unroll
            for (int j = 0; j < 4; ++j) {
                int pos = lsub + 32 * j;
                if (pos < HALO) {
                    int lg = l0 - PADL + pos;
                    float v = (lg >= 0 && lg < LLEN) ? xr[pos] : 0.f;
                    xs[pos][c] = f2bf(v);
                }
            }
        }
    }
    __syncthreads();

    const int mb0 = wave;          // M-block rows  wave*16 ..
    const int mb1 = wave + 8;      //               (wave+8)*16 ..
    v8f acc0[4] = {};              // [nb] for mb0
    v8f acc1[4] = {};              // [nb] for mb1

    for (int k = 0; k < KT; ++k) {
        // per-lane Gaussian splats for this tap (one per N-block)
        unsigned wsp[4];
        #pragma unroll
        for (int nb = 0; nb < 4; ++nb) wsp[nb] = wkb[k][nb * 16 + nIn];
        const int posBase = 6 * k + nIn;

        #pragma unroll 4
        for (int cc = 0; cc < 8; ++cc) {
            const int c0 = cc * 32;
            // A fragments from global (L2/WGP$-resident repacked weights)
            const unsigned short* pa0 =
                wbf + (mb0 * 16 + nIn) * KDIM + k * 256 + c0 + half * 8;
            const unsigned short* pa1 =
                wbf + (mb1 * 16 + nIn) * KDIM + k * 256 + c0 + half * 8;
            FragA a0, a1;
            a0.q[0] = *(const uint4*)(pa0);
            a0.q[1] = *(const uint4*)(pa0 + 16);
            a1.q[0] = *(const uint4*)(pa1);
            a1.q[1] = *(const uint4*)(pa1 + 16);

            #pragma unroll
            for (int nb = 0; nb < 4; ++nb) {
                const unsigned short* pb = &xs[posBase + nb * 16][c0 + half * 8];
                FragB f;
                f.q[0] = *(const uint4*)(pb);
                f.q[1] = *(const uint4*)(pb + 16);
                // scale the 16 bf16 activations by the per-lane Gaussian splat:
                // one VOP3P v_pk_mul_bf16 per dword (ISA 15.10, opcode 42)
                #pragma unroll
                for (int i = 0; i < 8; ++i)
                    asm("v_pk_mul_bf16 %0, %1, %2"
                        : "=v"(f.u[i]) : "v"(f.u[i]), "v"(wsp[nb]));

                acc0[nb] = __builtin_amdgcn_wmma_f32_16x16x32_bf16(
                    false, a0.v, false, f.v, (short)0, acc0[nb], false, false);
                acc1[nb] = __builtin_amdgcn_wmma_f32_16x16x32_bf16(
                    false, a1.v, false, f.v, (short)0, acc1[nb], false, false);
            }
        }
    }

    // ---- store pre-BN y (f32) per documented C-matrix layout ----
    #pragma unroll
    for (int nb = 0; nb < 4; ++nb) {
        int lcol = l0 + nb * 16 + nIn;
        #pragma unroll
        for (int r = 0; r < 8; ++r) {
            int o0 = mb0 * 16 + half * 8 + r;
            int o1 = mb1 * 16 + half * 8 + r;
            y[(b * COUT + o0) * LLEN + lcol] = acc0[nb][r];
            y[(b * COUT + o1) * LLEN + lcol] = acc1[nb][r];
        }
    }
}

// ---------------- 3) BN statistics: one block per channel -------------------
__global__ __launch_bounds__(256) void bn_stats(const float* __restrict__ y,
                                                float* __restrict__ stats) {
    const int o = blockIdx.x;
    const int t = threadIdx.x;
    float s = 0.f, s2 = 0.f;
    for (int b = 0; b < BATCH; ++b) {
        const float* p = y + (b * COUT + o) * LLEN;
        for (int i = t; i < LLEN; i += 256) {
            float v = p[i];
            s += v; s2 += v * v;
        }
    }
    __shared__ float rs[256], rq[256];
    rs[t] = s; rq[t] = s2;
    __syncthreads();
    for (int off = 128; off > 0; off >>= 1) {
        if (t < off) { rs[t] += rs[t + off]; rq[t] += rq[t + off]; }
        __syncthreads();
    }
    if (t == 0) {
        const float invN = 1.0f / (float)(BATCH * LLEN);
        float mean = rs[0] * invN;
        float var  = rq[0] * invN - mean * mean;
        stats[o]        = mean;
        stats[COUT + o] = rsqrtf(var + 1e-5f);
    }
}

// ---------------- 4) normalize + affine + ReLU (in place, float4) -----------
__global__ __launch_bounds__(256) void bn_apply(float* __restrict__ y,
                                                const float* __restrict__ stats,
                                                const float* __restrict__ gamma,
                                                const float* __restrict__ beta) {
    int idx = (blockIdx.x * 256 + threadIdx.x) * 4;    // row length 16384 % 4 == 0
    int o   = (idx >> 14) & 255;
    float m = stats[o];
    float r = stats[COUT + o];
    float g = gamma[o] * r;
    float bb = beta[o] - m * g;
    float4 v = *(float4*)(y + idx);
    v.x = fmaxf(v.x * g + bb, 0.f);
    v.y = fmaxf(v.y * g + bb, 0.f);
    v.z = fmaxf(v.z * g + bb, 0.f);
    v.w = fmaxf(v.w * g + bb, 0.f);
    *(float4*)(y + idx) = v;
}

// ---------------- host launch ----------------------------------------------
extern "C" void kernel_launch(void* const* d_in, const int* in_sizes, int n_in,
                              void* d_out, int out_size, void* d_ws, size_t ws_size,
                              hipStream_t stream) {
    const float* x      = (const float*)d_in[0];
    const float* coords = (const float*)d_in[1];
    const float* weight = (const float*)d_in[2];
    const float* gamma  = (const float*)d_in[3];
    const float* beta   = (const float*)d_in[4];
    float* y = (float*)d_out;

    unsigned short* wbf = (unsigned short*)d_ws;                    // 1,179,648 B
    float* stats = (float*)((char*)d_ws + (size_t)COUT * KDIM * 2); // mean[256], rstd[256]

    prep_w   <<<(COUT * KDIM) / 256, 256, 0, stream>>>(weight, wbf);
    conv_wmma<<<BATCH * (LLEN / LPB), 256, 0, stream>>>(x, coords, wbf, y);
    bn_stats <<<COUT, 256, 0, stream>>>(y, stats);
    bn_apply <<<NTOT / (256 * 4), 256, 0, stream>>>(y, stats, gamma, beta);
}
